// FixedLengthDecoder_48481590837459
// MI455X (gfx1250) — matrix operand order repaired and checked by reading, not compiled, not run on hardware
//
// MI455X (gfx1250, CDNA5) implementation of a 512-step LSTM decoder.
//
// Structure:
//   1. pack kernels : fp32 weights -> bf16, B-tile-major [K/32][N][32] layout
//                     (W_k and W_r fused into one 2048x4096 recurrent matrix)
//   2. embed_kernel : E = relu(X @ W_emb + b_emb)   (big parallel bf16 WMMA GEMM)
//   3. lstm_persistent_kernel : 512 sequential steps in ONE kernel with a
//                     device-scope sense-reversal grid barrier per step.
//                     z = [e_t | h] @ [W_k;W_r] + b_r ; gate math lane-local
//                     because each wave's 4 N-tiles are the 4 gates of the
//                     same 16 units. h double-buffered (bf16), c fp32.
//                     Inner K loop is software-pipelined one k-tile deep so
//                     WMMAs of tile kt overlap the loads of tile kt+1.
//   4. out_kernel   : Out = H @ W_out + b_out       (big parallel bf16 WMMA GEMM)
//
// All GEMMs use v_wmma_f32_16x16x32_bf16 (fp32 accumulate).
// Workspace requirement: ~291 MB (assumed available in d_ws).

#include <hip/hip_runtime.h>

typedef __attribute__((ext_vector_type(16))) __bf16 v16bf;
typedef __attribute__((ext_vector_type(8)))  __bf16 v8bf;
typedef __attribute__((ext_vector_type(8)))  float  v8f;
typedef __attribute__((ext_vector_type(4)))  float  v4f;

#define DEVINL __device__ __forceinline__

constexpr int BATCH  = 128;
constexpr int TSTEPS = 512;
constexpr int UNITS  = 1024;
constexpr int OUTD   = 1024;
constexpr int NGATE  = 4 * UNITS;            // 4096
constexpr int MROWS  = TSTEPS * BATCH;       // 65536 rows in E / H
constexpr int NWG_STEP = 32;                 // persistent-kernel grid

// ---------------- workspace layout (bytes) ----------------
constexpr size_t SZ_E    = (size_t)MROWS * UNITS * 2;        // bf16 E
constexpr size_t SZ_H    = (size_t)MROWS * UNITS * 2;        // bf16 H
constexpr size_t SZ_WKR  = (size_t)2 * UNITS * NGATE * 2;    // bf16 packed [W_k;W_r]
constexpr size_t SZ_WEMB = (size_t)OUTD * UNITS * 2;         // bf16 packed W_emb
constexpr size_t SZ_WOUT = (size_t)UNITS * OUTD * 2;         // bf16 packed W_out
constexpr size_t SZ_HBF  = (size_t)2 * BATCH * UNITS * 2;    // bf16 h double buffer
constexpr size_t SZ_C    = (size_t)BATCH * UNITS * 4;        // f32 c state

constexpr size_t OFF_E    = 0;
constexpr size_t OFF_H    = OFF_E    + SZ_E;
constexpr size_t OFF_WKR  = OFF_H    + SZ_H;
constexpr size_t OFF_WEMB = OFF_WKR  + SZ_WKR;
constexpr size_t OFF_WOUT = OFF_WEMB + SZ_WEMB;
constexpr size_t OFF_HBF  = OFF_WOUT + SZ_WOUT;
constexpr size_t OFF_C    = OFF_HBF  + SZ_HBF;
constexpr size_t OFF_SYNC = OFF_C    + SZ_C;

// ---------------- small helpers ----------------
DEVINL v8f zero8() {
  v8f z;
#pragma unroll
  for (int i = 0; i < 8; ++i) z[i] = 0.0f;
  return z;
}

DEVINL float sigm(float x) { return 1.0f / (1.0f + __expf(-x)); }

DEVINL float tanh_fast(float x) {
  float ax = fabsf(x);
  float e  = __expf(-2.0f * ax);
  float r  = (1.0f - e) / (1.0f + e);
  return copysignf(r, x);
}

DEVINL v8f wmma_bf16(v16bf a, v16bf b, v8f c) {
  // (neg_a, A, neg_b, B, c_mod, C, reuse_a, reuse_b)
  return __builtin_amdgcn_wmma_f32_16x16x32_bf16(false, a, false, b, (short)0, c,
                                                 false, false);
}

// A-matrix (16x32 bf16) per ISA layout: lane l (row = l%16) holds
//   element i : K = (i<8?0:16) + (l<16?0:8) + (i&7)
// -> two contiguous 8-half (16B) runs at row[k0 + (l<16?0:8)] and +16.
DEVINL v16bf load_a_bf16(const __bf16* __restrict__ row, int k0, int lane) {
  const __bf16* p = row + k0 + ((lane & 16) ? 8 : 0);
  v8bf lo = *(const v8bf*)(p);
  v8bf hi = *(const v8bf*)(p + 16);
  v16bf a;
#pragma unroll
  for (int i = 0; i < 8; ++i) { a[i] = lo[i]; a[8 + i] = hi[i]; }
  return a;
}

// Same A layout but converting fp32 source on the fly.
DEVINL v16bf load_a_f32(const float* __restrict__ row, int k0, int lane) {
  const float* p = row + k0 + ((lane & 16) ? 8 : 0);
  v4f f0 = *(const v4f*)(p);
  v4f f1 = *(const v4f*)(p + 4);
  v4f f2 = *(const v4f*)(p + 16);
  v4f f3 = *(const v4f*)(p + 20);
  v16bf a;
#pragma unroll
  for (int i = 0; i < 4; ++i) {
    a[i]      = (__bf16)f0[i];
    a[4 + i]  = (__bf16)f1[i];
    a[8 + i]  = (__bf16)f2[i];
    a[12 + i] = (__bf16)f3[i];
  }
  return a;
}

// B-matrix (32x16 bf16): lane l (col = l%16) holds element i:
//   K = k0 + (i<8?0:16) + (l<16?0:8) + (i&7)
// Packed weight layout pack[(kt*N + n)*32 + kk] = W[kt*32+kk][n]
// -> two contiguous 16B loads per lane.
DEVINL v16bf load_b_pack(const __bf16* __restrict__ pack, int N, int kt, int n0,
                         int lane) {
  const __bf16* p =
      pack + ((size_t)kt * N + (n0 + (lane & 15))) * 32 + ((lane & 16) ? 8 : 0);
  v8bf lo = *(const v8bf*)(p);
  v8bf hi = *(const v8bf*)(p + 16);
  v16bf b;
#pragma unroll
  for (int i = 0; i < 8; ++i) { b[i] = lo[i]; b[8 + i] = hi[i]; }
  return b;
}

// Device-scope sense-reversal grid barrier.
DEVINL void grid_barrier(unsigned* cnt, unsigned* gen, unsigned nwg) {
  __threadfence();          // make this block's stores device-visible
  __syncthreads();
  if (threadIdx.x == 0) {
    unsigned g = __hip_atomic_load(gen, __ATOMIC_RELAXED, __HIP_MEMORY_SCOPE_AGENT);
    unsigned arrived =
        __hip_atomic_fetch_add(cnt, 1u, __ATOMIC_ACQ_REL, __HIP_MEMORY_SCOPE_AGENT) + 1;
    if (arrived == nwg) {
      __hip_atomic_store(cnt, 0u, __ATOMIC_RELAXED, __HIP_MEMORY_SCOPE_AGENT);
      __hip_atomic_store(gen, g + 1u, __ATOMIC_RELEASE, __HIP_MEMORY_SCOPE_AGENT);
    } else {
      while (__hip_atomic_load(gen, __ATOMIC_ACQUIRE, __HIP_MEMORY_SCOPE_AGENT) == g) {
        __builtin_amdgcn_s_sleep(2);
      }
    }
  }
  __syncthreads();
}

// ---------------- prep kernels ----------------
__global__ void init_state_kernel(const float* __restrict__ h0,
                                  const float* __restrict__ c0,
                                  __bf16* __restrict__ hbuf,
                                  float* __restrict__ c_state,
                                  unsigned* __restrict__ sync) {
  int idx = blockIdx.x * 256 + threadIdx.x;          // 131072 total
  hbuf[idx]    = (__bf16)h0[idx];                    // buffer 0
  c_state[idx] = c0[idx];
  if (idx < 2) sync[idx] = 0u;
}

// Generic fp32 [K][N] -> bf16 B-tile-major pack. Launch with K*N/256 blocks.
__global__ void pack_w_kernel(const float* __restrict__ src, __bf16* __restrict__ dst,
                              int N) {
  int idx = blockIdx.x * 256 + threadIdx.x;
  int kk = idx & 31;
  int n  = (idx >> 5) % N;
  int kt = (idx >> 5) / N;
  dst[idx] = (__bf16)src[(size_t)(kt * 32 + kk) * N + n];
}

// Fused [W_k ; W_r] (2048 x 4096) pack.
__global__ void pack_wkr_kernel(const float* __restrict__ Wk,
                                const float* __restrict__ Wr,
                                __bf16* __restrict__ dst) {
  int idx = blockIdx.x * 256 + threadIdx.x;          // 2048*4096 total
  int kk = idx & 31;
  int n  = (idx >> 5) & 4095;
  int kt = idx >> 17;
  int k  = kt * 32 + kk;
  float v = (k < UNITS) ? Wk[(size_t)k * NGATE + n]
                        : Wr[(size_t)(k - UNITS) * NGATE + n];
  dst[idx] = (__bf16)v;
}

// ---------------- E = relu(X @ W_emb + b_emb) ----------------
// grid 8192 x 256; WG tile 128 rows x 64 cols; wave tile 16x64.
__global__ void embed_kernel(const float* __restrict__ X,
                             const __bf16* __restrict__ Wp,
                             const float* __restrict__ b_emb,
                             __bf16* __restrict__ E) {
  int wg = blockIdx.x;
  int mg = wg >> 4, ng = wg & 15;
  int wave = threadIdx.x >> 5, lane = threadIdx.x & 31;
  int row0 = mg * 128 + wave * 16;
  int rA   = row0 + (lane & 15);
  int t = rA >> 7, b = rA & 127;
  const float* xrow = X + ((size_t)b * (TSTEPS + 1) + t) * OUTD;
  int n0 = ng * 64;

  v8f acc[4];
#pragma unroll
  for (int j = 0; j < 4; ++j) acc[j] = zero8();

  for (int kt = 0; kt < OUTD / 32; ++kt) {
    // all loads first (fresh registers), then the 4 WMMAs back-to-back
    v16bf a = load_a_f32(xrow, kt * 32, lane);
    v16bf bm[4];
#pragma unroll
    for (int j = 0; j < 4; ++j)
      bm[j] = load_b_pack(Wp, UNITS, kt, n0 + 16 * j, lane);
#pragma unroll
    for (int j = 0; j < 4; ++j) acc[j] = wmma_bf16(a, bm[j], acc[j]);
  }

  int rbase = row0 + ((lane & 16) ? 8 : 0);
#pragma unroll
  for (int j = 0; j < 4; ++j) {
    int col = n0 + 16 * j + (lane & 15);
    float bias = b_emb[col];
#pragma unroll
    for (int v = 0; v < 8; ++v) {
      float e = acc[j][v] + bias;
      e = e > 0.0f ? e : 0.0f;
      E[(size_t)(rbase + v) * UNITS + col] = (__bf16)e;
    }
  }
}

// ---------------- persistent recurrent kernel ----------------
// grid 32 x 256. WG w owns units [32w, 32w+32). 8 waves = 4(M) x 2(unit-half):
// wave tile = 32 rows x (16 units x 4 gates). Gate tiles of one unit block land
// in the same lane -> lane-local LSTM cell math.
struct Frag {
  v16bf a0, a1;   // two 16x32 A sub-tiles (rows rA0, rA0+16)
  v16bf b[4];     // one 32x16 B tile per gate
};

DEVINL Frag load_frag(const __bf16* __restrict__ A, const __bf16* __restrict__ Wkr,
                      int rA0, int k0, int ktB, int uu0, int lane) {
  Frag f;
  f.a0 = load_a_bf16(A + (size_t)rA0 * UNITS,        k0, lane);
  f.a1 = load_a_bf16(A + (size_t)(rA0 + 16) * UNITS, k0, lane);
#pragma unroll
  for (int g = 0; g < 4; ++g)
    f.b[g] = load_b_pack(Wkr, NGATE, ktB, g * UNITS + uu0, lane);
  return f;
}

__global__ void lstm_persistent_kernel(const __bf16* __restrict__ E,
                                       const __bf16* __restrict__ Wkr,
                                       const float* __restrict__ b_r,
                                       float* __restrict__ c_state,
                                       __bf16* __restrict__ hbuf,   // 2 buffers
                                       __bf16* __restrict__ H,
                                       unsigned* __restrict__ sync) {
  int w    = blockIdx.x;
  int wave = threadIdx.x >> 5, lane = threadIdx.x & 31;
  int mi = wave >> 1, ni = wave & 1;
  int row0  = mi * 32;                      // rows [row0, row0+32)
  int uu0   = w * 32 + ni * 16;             // units [uu0, uu0+16)
  int u     = uu0 + (lane & 15);
  int rA0   = row0 + (lane & 15);           // A row for sub-tile 0 (+16 for 1)
  int rbase = row0 + ((lane & 16) ? 8 : 0); // C row base: + 16a + v

  float bias[4];
#pragma unroll
  for (int g = 0; g < 4; ++g) bias[g] = b_r[g * UNITS + u];

  unsigned* cnt = sync;
  unsigned* gen = sync + 1;

#pragma unroll 1
  for (int t = 0; t < TSTEPS; ++t) {
    const __bf16* A1   = E + (size_t)t * (BATCH * UNITS);       // K 0..1023
    const __bf16* A2   = hbuf + (size_t)(t & 1) * (BATCH * UNITS); // K 1024..2047
    __bf16*       hout = hbuf + (size_t)((t + 1) & 1) * (BATCH * UNITS);

    v8f acc[2][4];
#pragma unroll
    for (int a = 0; a < 2; ++a)
#pragma unroll
      for (int g = 0; g < 4; ++g) acc[a][g] = zero8();

    // Unified 64-tile K loop over [e_t | h], software-pipelined 1 deep:
    // loads of tile kt+1 are in flight while the 8 WMMAs of tile kt execute.
    Frag f = load_frag(A1, Wkr, rA0, 0, 0, uu0, lane);
#pragma unroll 2
    for (int kt = 0; kt < 64; ++kt) {
      Frag fn;
      if (kt < 63) {
        int ktn = kt + 1;
        const __bf16* An = (ktn < 32) ? A1 : A2;
        fn = load_frag(An, Wkr, rA0, (ktn & 31) * 32, ktn, uu0, lane);
      }
#pragma unroll
      for (int g = 0; g < 4; ++g) acc[0][g] = wmma_bf16(f.a0, f.b[g], acc[0][g]);
#pragma unroll
      for (int g = 0; g < 4; ++g) acc[1][g] = wmma_bf16(f.a1, f.b[g], acc[1][g]);
      f = fn;
    }

    // Lane-local LSTM cell update (zi, zf, zg, zo all in this lane).
#pragma unroll
    for (int a = 0; a < 2; ++a) {
#pragma unroll
      for (int v = 0; v < 8; ++v) {
        int row = rbase + 16 * a + v;
        size_t idx = (size_t)row * UNITS + u;
        float zi = acc[a][0][v] + bias[0];
        float zf = acc[a][1][v] + bias[1];
        float zg = acc[a][2][v] + bias[2];
        float zo = acc[a][3][v] + bias[3];
        float cold = c_state[idx];
        float cn = sigm(zf) * cold + sigm(zi) * tanh_fast(zg);
        float hn = sigm(zo) * tanh_fast(cn);
        c_state[idx] = cn;
        __bf16 hb = (__bf16)hn;
        hout[idx] = hb;
        H[(size_t)t * (BATCH * UNITS) + idx] = hb;
      }
    }

    grid_barrier(cnt, gen, NWG_STEP);
  }
}

// ---------------- Out = H @ W_out + b_out ----------------
__global__ void out_kernel(const __bf16* __restrict__ H,
                           const __bf16* __restrict__ Wp,
                           const float* __restrict__ b_out,
                           float* __restrict__ out) {
  int wg = blockIdx.x;
  int mg = wg >> 4, ng = wg & 15;
  int wave = threadIdx.x >> 5, lane = threadIdx.x & 31;
  int row0 = mg * 128 + wave * 16;
  int rA   = row0 + (lane & 15);
  const __bf16* hrow = H + (size_t)rA * UNITS;
  int n0 = ng * 64;

  v8f acc[4];
#pragma unroll
  for (int j = 0; j < 4; ++j) acc[j] = zero8();

  for (int kt = 0; kt < UNITS / 32; ++kt) {
    v16bf a = load_a_bf16(hrow, kt * 32, lane);
    v16bf bm[4];
#pragma unroll
    for (int j = 0; j < 4; ++j)
      bm[j] = load_b_pack(Wp, OUTD, kt, n0 + 16 * j, lane);
#pragma unroll
    for (int j = 0; j < 4; ++j) acc[j] = wmma_bf16(a, bm[j], acc[j]);
  }

  int rbase = row0 + ((lane & 16) ? 8 : 0);
#pragma unroll
  for (int j = 0; j < 4; ++j) {
    int col = n0 + 16 * j + (lane & 15);
    float bias = b_out[col];
#pragma unroll
    for (int v = 0; v < 8; ++v) {
      int row = rbase + v;              // = t*128 + b
      int t = row >> 7, b = row & 127;
      out[((size_t)b * TSTEPS + t) * OUTD + col] = acc[j][v] + bias;
    }
  }
}

// ---------------- host side ----------------
extern "C" void kernel_launch(void* const* d_in, const int* in_sizes, int n_in,
                              void* d_out, int out_size, void* d_ws, size_t ws_size,
                              hipStream_t stream) {
  const float* h0    = (const float*)d_in[0];
  const float* c0    = (const float*)d_in[1];
  const float* X     = (const float*)d_in[2];   // [B][T+1][OUT]
  const float* W_emb = (const float*)d_in[3];
  const float* b_emb = (const float*)d_in[4];
  const float* W_k   = (const float*)d_in[5];
  const float* W_r   = (const float*)d_in[6];
  const float* b_r   = (const float*)d_in[7];
  const float* W_out = (const float*)d_in[8];
  const float* b_out = (const float*)d_in[9];
  // d_in[10] = seq_len (== TSTEPS, compiled in)

  char* ws = (char*)d_ws;
  __bf16*   E      = (__bf16*)(ws + OFF_E);
  __bf16*   H      = (__bf16*)(ws + OFF_H);
  __bf16*   WkrP   = (__bf16*)(ws + OFF_WKR);
  __bf16*   WembP  = (__bf16*)(ws + OFF_WEMB);
  __bf16*   WoutP  = (__bf16*)(ws + OFF_WOUT);
  __bf16*   hbuf   = (__bf16*)(ws + OFF_HBF);
  float*    cstate = (float*)(ws + OFF_C);
  unsigned* sync   = (unsigned*)(ws + OFF_SYNC);
  float*    out    = (float*)d_out;

  init_state_kernel<<<dim3(BATCH * UNITS / 256), dim3(256), 0, stream>>>(
      h0, c0, hbuf, cstate, sync);
  pack_w_kernel<<<dim3(OUTD * UNITS / 256), dim3(256), 0, stream>>>(W_emb, WembP, UNITS);
  pack_wkr_kernel<<<dim3(2 * UNITS * NGATE / 256), dim3(256), 0, stream>>>(W_k, W_r, WkrP);
  pack_w_kernel<<<dim3(UNITS * OUTD / 256), dim3(256), 0, stream>>>(W_out, WoutP, OUTD);

  embed_kernel<<<dim3((MROWS / 128) * (UNITS / 64)), dim3(256), 0, stream>>>(
      X, WembP, b_emb, E);

  lstm_persistent_kernel<<<dim3(NWG_STEP), dim3(256), 0, stream>>>(
      E, WkrP, b_r, cstate, hbuf, H, sync);

  out_kernel<<<dim3((MROWS / 128) * (OUTD / 64)), dim3(256), 0, stream>>>(
      H, WoutP, b_out, out);
}
